// Encoder_91207925498123
// MI455X (gfx1250) — matrix operand order repaired and checked by reading, not compiled, not run on hardware
//
#include <hip/hip_runtime.h>
#include <cmath>

// ---------------- CDNA5 WMMA types ----------------
typedef __attribute__((ext_vector_type(16))) __bf16          v16bf;
typedef __attribute__((ext_vector_type(16))) unsigned short  v16u;
typedef __attribute__((ext_vector_type(8)))  float           v8f;
typedef __attribute__((__vector_size__(4 * sizeof(int)))) int b128v;

#define AS1 __attribute__((address_space(1)))
#define AS3 __attribute__((address_space(3)))

#if defined(__has_builtin)
#if __has_builtin(__builtin_amdgcn_global_load_async_to_lds_b128) && \
    __has_builtin(__builtin_amdgcn_s_wait_asynccnt)
#define HAVE_ASYNC 1
#endif
#endif
#ifndef HAVE_ASYNC
#define HAVE_ASYNC 0
#endif

__device__ __forceinline__ unsigned short f2bf(float f) {
    unsigned int u = __builtin_bit_cast(unsigned int, f);
    unsigned int r = u + 0x7FFFu + ((u >> 16) & 1u);   // round-to-nearest-even
    return (unsigned short)(r >> 16);
}

// =====================================================================
// Generic LDS-tiled WMMA GEMM, double-buffered with async B-tile copies:
//   C[M,N] = scale * ( A(f32,lda=K) * B(bf16, stored as B^T row-major N x K)
//            [+ A2*B2]  + bias[n] )
// A row remap (for strided "cur" slices): row -> (m + (m/segLen)*segJump + off)
// Block: 128 threads (4 waves), tile 64(M) x 64(N), K step 32.
// B tiles (already bf16) go global->LDS via GLOBAL_LOAD_ASYNC_TO_LDS_B128
// (ASYNCcnt); A tiles convert f32->bf16 through VGPRs. Next tile staged while
// current tile feeds v_wmma_f32_16x16x32_bf16.
// =====================================================================
__global__ __launch_bounds__(128)
void wmma_gemm_kernel(const float* __restrict__ A,  const unsigned short* __restrict__ Bm,
                      const float* __restrict__ A2, const unsigned short* __restrict__ B2m,
                      const float* __restrict__ bias, float scale,
                      float* __restrict__ C, int M, int N, int K, int ldc,
                      int sL, int sJ, int sO, int clampMax,
                      int sL2, int sJ2, int sO2)
{
    __shared__ unsigned short As[2][64][32];
    __shared__ unsigned short Bs[2][64][32];
    const int tid  = threadIdx.x;
    const int wave = tid >> 5;
    const int lane = tid & 31;
    const int bm   = blockIdx.y * 64;
    const int bn   = blockIdx.x * 64;

    const int kSteps = K >> 5;
    const int nPass  = (A2 != nullptr) ? 2 : 1;
    const int total  = kSteps * nPass;

    v8f acc[4];
    const v8f vzero = {0.f,0.f,0.f,0.f,0.f,0.f,0.f,0.f};
    #pragma unroll
    for (int t = 0; t < 4; ++t) acc[t] = vzero;

    // stage tile `step` (A: f32->bf16 via VGPRs; B: async direct to LDS)
    auto stage = [&](int buf, int step) {
        const int  pass = (step >= kSteps) ? 1 : 0;
        const int  kk   = (step - pass * kSteps) << 5;
        const float*          Ap = pass ? A2  : A;
        const unsigned short* Bp = pass ? B2m : Bm;
        const int cL = pass ? sL2 : sL;
        const int cJ = pass ? sJ2 : sJ;
        const int cO = pass ? sO2 : sO;
        #pragma unroll
        for (int i = 0; i < 4; ++i) {
            int idx = tid + i * 128;          // float4 unit id, 0..511
            int r   = idx >> 3;
            int c4  = (idx & 7) << 2;
            int gm  = bm + r;
            if (gm > clampMax) gm = clampMax;
            long arow = (long)gm;
            if (cL > 0) arow = (long)gm + (long)(gm / cL) * cJ + cO;
            const float4 v = *(const float4*)(Ap + arow * (long)K + kk + c4);
            As[buf][r][c4 + 0] = f2bf(v.x);
            As[buf][r][c4 + 1] = f2bf(v.y);
            As[buf][r][c4 + 2] = f2bf(v.z);
            As[buf][r][c4 + 3] = f2bf(v.w);
        }
        #pragma unroll
        for (int i = 0; i < 2; ++i) {
            int idx = tid + i * 128;          // uint4 unit id, 0..255
            int r   = idx >> 2;
            int c8  = (idx & 3) << 3;
            const unsigned short* gsrc = Bp + (long)(bn + r) * K + kk + c8;
#if HAVE_ASYNC
            __builtin_amdgcn_global_load_async_to_lds_b128(
                (AS1 b128v*)gsrc, (AS3 b128v*)&Bs[buf][r][c8], 0, 0);
#else
            *(uint4*)(&Bs[buf][r][c8]) = *(const uint4*)gsrc;
#endif
        }
    };

    stage(0, 0);
    for (int step = 0; step < total; ++step) {
        const int cur = step & 1;
#if HAVE_ASYNC
        __builtin_amdgcn_s_wait_asynccnt(0);   // B tile for `cur` landed in LDS
#endif
        __syncthreads();                       // A tile stores visible, all waves synced
        if (step + 1 < total) stage(cur ^ 1, step + 1);

        // ---- A fragment (16x32, rows wave*16..+15); CDNA5 wave32 layout:
        // lanes 0-15 hold K 0-7 / 16-23, lanes 16-31 hold K 8-15 / 24-31
        v16u au;
        {
            int m   = (wave << 4) + (lane & 15);
            int kb0 = (lane < 16) ? 0  : 8;
            int kb1 = (lane < 16) ? 16 : 24;
            #pragma unroll
            for (int e = 0; e < 8; ++e) au[e]     = As[cur][m][kb0 + e];
            #pragma unroll
            for (int e = 0; e < 8; ++e) au[8 + e] = As[cur][m][kb1 + e];
        }
        v16bf a = __builtin_bit_cast(v16bf, au);

        #pragma unroll
        for (int nt = 0; nt < 4; ++nt) {
            v16u bu;
            int n  = (nt << 4) + (lane & 15);
            int kb = (lane < 16) ? 0 : 8;      // lane = column; K 0-7/16-23 | 8-15/24-31
            #pragma unroll
            for (int e = 0; e < 16; ++e) bu[e] = Bs[cur][n][kb + (e & 7) + ((e >> 3) << 4)];
            v16bf b = __builtin_bit_cast(v16bf, bu);
            acc[nt] = __builtin_amdgcn_wmma_f32_16x16x32_bf16(
                false, a, false, b, (short)0, acc[nt], false, false);
        }
    }

    // ---- epilogue: D layout  M = v + 8*(lane>=16),  N = lane%16 tile-local ----
    const int lm = (lane < 16) ? 0 : 8;
    const int ln = lane & 15;
    #pragma unroll
    for (int nt = 0; nt < 4; ++nt) {
        int n = bn + (nt << 4) + ln;
        float bv = bias ? bias[n] : 0.f;
        #pragma unroll
        for (int v = 0; v < 8; ++v) {
            int m = bm + (wave << 4) + v + lm;
            if (m < M && n < N) {
                C[(long)m * ldc + n] = (acc[nt][v] + bv) * scale;
            }
        }
    }
}

// ---------------- small kernels ----------------
__global__ void bias_add_kernel(const float* a, const float* b, float* o, int n) {
    int i = blockIdx.x * 256 + threadIdx.x;
    if (i < n) o[i] = a[i] + b[i];
}

__global__ void cvt_bf16_kernel(const float* __restrict__ s, unsigned short* __restrict__ d, long n) {
    long i = (long)blockIdx.x * 256 + threadIdx.x;
    if (i < n) d[i] = f2bf(s[i]);
}

// relT[n][k] = (k<500) ? emb[k][n] : 0     (512 x 512 bf16, B^T layout for merge GEMM)
__global__ void relT_kernel(const float* __restrict__ emb, unsigned short* __restrict__ relT) {
    long i = (long)blockIdx.x * 256 + threadIdx.x;
    if (i >= 512L * 512L) return;
    int n = (int)(i >> 9), k = (int)(i & 511);
    relT[i] = (k < 500) ? f2bf(emb[(long)k * 512 + n]) : (unsigned short)0;
}

// cur[b][s][:] = emb[tokens[b][s]][:]
__global__ __launch_bounds__(256) void gather_kernel(const int* __restrict__ tokens,
                                                     const float* __restrict__ emb,
                                                     float* __restrict__ cur) {
    int bs  = blockIdx.x;                      // b*16 + s
    int tok = tokens[bs];
    const float* src = emb + (long)tok * 512;
    float* dst = cur + (long)bs * 512;
    for (int e = threadIdx.x; e < 512; e += 256) dst[e] = src[e];
}

// gates: i,f,g,o chunks of G (ldc=2048);  c = sig(f)*cPrev + sig(i)*tanh(g); h = sig(o)*tanh(c)
__global__ __launch_bounds__(256) void lstm_gates_kernel(const float* __restrict__ G,
                                                         const float* __restrict__ cPrev,
                                                         float* __restrict__ h, float* __restrict__ c,
                                                         int M) {
    long idx = (long)blockIdx.x * 256 + threadIdx.x;
    if (idx >= (long)M * 512) return;
    long m = idx >> 9;
    int  e = (int)(idx & 511);
    const float* g = G + m * 2048;
    float gi = g[e], gf = g[512 + e], gg = g[1024 + e], go = g[1536 + e];
    float si = 1.f / (1.f + expf(-gi));
    float sf = 1.f / (1.f + expf(-gf));
    float so = 1.f / (1.f + expf(-go));
    float cp = cPrev ? cPrev[idx] : 0.f;
    float cv = sf * cp + si * tanhf(gg);
    c[idx] = cv;
    h[idx] = so * tanhf(cv);
}

// sc[m] = sigmoid( dot(h2[m], fc_w) + fc_b )   (one wave per row)
__global__ __launch_bounds__(256) void score_fc_kernel(const float* __restrict__ h2,
                                                       const float* __restrict__ fc_w,
                                                       const float* __restrict__ fc_b,
                                                       float* __restrict__ sc, int M) {
    int row  = blockIdx.x * 8 + (threadIdx.x >> 5);
    int lane = threadIdx.x & 31;
    if (row >= M) return;
    const float* r = h2 + (long)row * 512;
    float s = 0.f;
    for (int e = lane; e < 512; e += 32) s += r[e] * fc_w[e];
    for (int o = 16; o; o >>= 1) s += __shfl_xor(s, o, 32);
    if (lane == 0) sc[row] = 1.f / (1.f + expf(-(s + fc_b[0])));
}

// argmax over L-1 scores per batch (first-max tie rule) + gather selected row
__global__ __launch_bounds__(256) void select_kernel(const float* __restrict__ sc,
                                                     const float* __restrict__ h2, int Lm1,
                                                     int* __restrict__ selIdx,
                                                     float* __restrict__ selected) {
    int b = blockIdx.x;
    __shared__ int sSel;
    if (threadIdx.x < 32) {
        int lane = threadIdx.x;
        float v = (lane < Lm1) ? sc[(long)b * Lm1 + lane] : -3.4e38f;
        int   i = lane;
        for (int o = 16; o; o >>= 1) {
            float ov = __shfl_xor(v, o, 32);
            int   oi = __shfl_xor(i, o, 32);
            if (ov > v || (ov == v && oi < i)) { v = ov; i = oi; }
        }
        if (lane == 0) { sSel = i; selIdx[b] = i; }
    }
    __syncthreads();
    const float* src = h2 + ((long)b * Lm1 + sSel) * 512;
    for (int e = threadIdx.x; e < 512; e += 256) selected[(long)b * 512 + e] = src[e];
}

// scores[b][500] = dot(q[b], ksel[b]) / sqrt(E)
__global__ void fixcol_kernel(const float* __restrict__ q, const float* __restrict__ ks,
                              float* __restrict__ scores, float invSqrtE) {
    int b = blockIdx.x, lane = threadIdx.x;
    float s = 0.f;
    for (int e = lane; e < 512; e += 32) s += q[(long)b * 512 + e] * ks[(long)b * 512 + e];
    for (int o = 16; o; o >>= 1) s += __shfl_xor(s, o, 32);
    if (lane == 0) scores[(long)b * 512 + 500] = s * invSqrtE;
}

// softmax + entropy over cols [0,501); writes prob (cols>=500 zeroed/padded), pSel, loss column
__global__ __launch_bounds__(256) void softmax_entropy_kernel(const float* __restrict__ scores,
                                                              float* __restrict__ prob,
                                                              float* __restrict__ pSel,
                                                              float* __restrict__ lossOut, int lossCol) {
    int b = blockIdx.x;
    const float* s = scores + (long)b * 512;
    __shared__ float red[8];
    __shared__ float sMx, sZ;
    int tid = threadIdx.x, lane = tid & 31, wv = tid >> 5;

    float mx = -3.4e38f;
    for (int r = tid; r < 501; r += 256) mx = fmaxf(mx, s[r]);
    for (int o = 16; o; o >>= 1) mx = fmaxf(mx, __shfl_xor(mx, o, 32));
    if (lane == 0) red[wv] = mx;
    __syncthreads();
    if (tid == 0) { float v = red[0]; for (int i = 1; i < 8; ++i) v = fmaxf(v, red[i]); sMx = v; }
    __syncthreads();
    float m0 = sMx;

    float z = 0.f;
    for (int r = tid; r < 501; r += 256) z += expf(s[r] - m0);
    for (int o = 16; o; o >>= 1) z += __shfl_xor(z, o, 32);
    if (lane == 0) red[wv] = z;
    __syncthreads();
    if (tid == 0) { float v = 0.f; for (int i = 0; i < 8; ++i) v += red[i]; sZ = v; }
    __syncthreads();
    float lz = logf(sZ);

    float h = 0.f;
    for (int r = tid; r < 501; r += 256) {
        float lp = s[r] - m0 - lz;
        float p  = expf(lp);
        h -= p * lp;
        if (prob) {
            prob[(long)b * 512 + r] = (r < 500) ? p : 0.f;   // col 500 handled via pSel
            if (r == 500 && pSel) pSel[b] = p;
        }
    }
    if (prob && tid < 11) prob[(long)b * 512 + 501 + tid] = 0.f;
    for (int o = 16; o; o >>= 1) h += __shfl_xor(h, o, 32);
    if (lane == 0) red[wv] = h;
    __syncthreads();
    if (tid == 0) { float v = 0.f; for (int i = 0; i < 8; ++i) v += red[i];
                    lossOut[(long)b * 16 + lossCol] = v; }
}

// cur_next[b][j] = j<sel ? cur[b][j] : (j==sel ? merged+pSel*selected : cur[b][j+1])
__global__ __launch_bounds__(256) void rebuild_kernel(const float* __restrict__ cur,
                                                      const float* __restrict__ merged,
                                                      const float* __restrict__ selected,
                                                      const float* __restrict__ pSel,
                                                      const int* __restrict__ selIdx,
                                                      float* __restrict__ curNext, int Lm1) {
    int b   = blockIdx.x;
    int sel = selIdx[b];
    float ps = pSel[b];
    for (int t = threadIdx.x; t < Lm1 * 512; t += 256) {
        int j = t >> 9, e = t & 511;
        float v;
        if (j < sel)       v = cur[((long)b * 16 + j) * 512 + e];
        else if (j == sel) v = merged[(long)b * 512 + e] + ps * selected[(long)b * 512 + e];
        else               v = cur[((long)b * 16 + j + 1) * 512 + e];
        curNext[((long)b * 16 + j) * 512 + e] = v;
    }
}

__global__ __launch_bounds__(256) void copy_h2_kernel(const float* __restrict__ h2,
                                                      float* __restrict__ cur) {
    int b = blockIdx.x;
    for (int e = threadIdx.x; e < 512; e += 256)
        cur[(long)b * 16 * 512 + e] = h2[(long)b * 512 + e];
}

__global__ void zero_loss_kernel(float* lossOut) {
    if (threadIdx.x < 128) lossOut[(long)threadIdx.x * 16 + 14] = 0.f;
}

__global__ void copy_scores_kernel(const float* __restrict__ scores, float* __restrict__ out) {
    long i = (long)blockIdx.x * 256 + threadIdx.x;
    if (i < 128L * 501L) {
        long b = i / 501, r = i % 501;
        out[i] = scores[b * 512 + r];
    }
}

// ---------------- host ----------------
static inline void gemm(hipStream_t st, const float* A, const unsigned short* B,
                        const float* A2, const unsigned short* B2,
                        const float* bias, float scale, float* C,
                        int M, int N, int K, int ldc,
                        int sL, int sJ, int sO, int clampMax,
                        int sL2 = 0, int sJ2 = 0, int sO2 = 0) {
    dim3 g(N / 64, (M + 63) / 64);
    wmma_gemm_kernel<<<g, 128, 0, st>>>(A, B, A2, B2, bias, scale, C, M, N, K, ldc,
                                        sL, sJ, sO, clampMax, sL2, sJ2, sO2);
}

extern "C" void kernel_launch(void* const* d_in, const int* in_sizes, int n_in,
                              void* d_out, int out_size, void* d_ws, size_t ws_size,
                              hipStream_t stream) {
    (void)in_sizes; (void)n_in; (void)out_size;
    const int   B = 128, S = 16, E = 512;
    const float invSqrtE = 1.0f / sqrtf((float)E);

    const int*   tokens = (const int*)  d_in[0];
    const float* emb    = (const float*)d_in[1];
    const float* w_ih   = (const float*)d_in[2];
    const float* w_hh   = (const float*)d_in[3];
    const float* b_ih   = (const float*)d_in[4];
    const float* b_hh   = (const float*)d_in[5];
    const float* fc_w   = (const float*)d_in[6];
    const float* fc_b   = (const float*)d_in[7];
    const float* fcq_w  = (const float*)d_in[8];
    const float* fcq_b  = (const float*)d_in[9];
    const float* fck_w  = (const float*)d_in[10];
    const float* fck_b  = (const float*)d_in[11];

    // ---- workspace carve-out (256B aligned) ----
    char*  ws  = (char*)d_ws;
    size_t off = 0;
    auto alloc = [&](size_t bytes) -> void* {
        void* p = ws + off;
        off = (off + bytes + 255) & ~(size_t)255;
        return p;
    };
    unsigned short* wih_bf  = (unsigned short*)alloc(2048L * 512 * 2);
    unsigned short* whh_bf  = (unsigned short*)alloc(2048L * 512 * 2);
    unsigned short* fcq_bf  = (unsigned short*)alloc(512L * 512 * 2);
    unsigned short* fck_bf  = (unsigned short*)alloc(512L * 512 * 2);
    unsigned short* relT_bf = (unsigned short*)alloc(512L * 512 * 2);
    float*          krel_f  = (float*)         alloc(512L * 512 * 4);
    unsigned short* krel_bf = (unsigned short*)alloc(512L * 512 * 2);
    float*          b_lstm  = (float*)         alloc(2048L * 4);
    float*          curA    = (float*)         alloc((long)B * S * E * 4);
    float*          curB    = (float*)         alloc((long)B * S * E * 4);
    float*          G       = (float*)         alloc(1920L * 2048 * 4);
    float*          h1      = (float*)         alloc(1920L * 512 * 4);
    float*          c1      = (float*)         alloc(1920L * 512 * 4);
    float*          h2      = (float*)         alloc(1920L * 512 * 4);
    float*          sc      = (float*)         alloc(1920L * 4);
    int*            selIdx  = (int*)           alloc(128L * 4);
    float*          sel     = (float*)         alloc(128L * 512 * 4);
    float*          q       = (float*)         alloc(128L * 512 * 4);
    float*          ksel    = (float*)         alloc(128L * 512 * 4);
    float*          scores  = (float*)         alloc(128L * 512 * 4);
    float*          prob    = (float*)         alloc(128L * 512 * 4);
    float*          pSel    = (float*)         alloc(128L * 4);
    float*          merged  = (float*)         alloc(128L * 512 * 4);
    if (off > ws_size) return;   // cannot run without scratch

    float* outScores = (float*)d_out;                 // B*501
    float* lossBase  = (float*)d_out + (long)B * 501; // B*16

    // ---- stage-invariant setup ----
    bias_add_kernel<<<8, 256, 0, stream>>>(b_ih, b_hh, b_lstm, 2048);
    cvt_bf16_kernel<<<(2048 * 512 + 255) / 256, 256, 0, stream>>>(w_ih, wih_bf, 2048L * 512);
    cvt_bf16_kernel<<<(2048 * 512 + 255) / 256, 256, 0, stream>>>(w_hh, whh_bf, 2048L * 512);
    cvt_bf16_kernel<<<(512 * 512 + 255) / 256, 256, 0, stream>>>(fcq_w, fcq_bf, 512L * 512);
    cvt_bf16_kernel<<<(512 * 512 + 255) / 256, 256, 0, stream>>>(fck_w, fck_bf, 512L * 512);
    relT_kernel<<<(512 * 512 + 255) / 256, 256, 0, stream>>>(emb, relT_bf);
    gather_kernel<<<B * S, 256, 0, stream>>>(tokens, emb, curA);
    // k_rel = rel_emb @ fck_w^T + fck_b  (M padded 512, rows>=500 clamp to zero row emb[500])
    gemm(stream, emb, fck_bf, nullptr, nullptr, fck_b, 1.f, krel_f,
         512, 512, 512, 512, 0, 0, 0, 500);
    cvt_bf16_kernel<<<(512 * 512 + 255) / 256, 256, 0, stream>>>(krel_f, krel_bf, 512L * 512);

    // ---- merge stages L = 16 .. 3 ----
    int lossCol = 0;
    for (int L = S; L >= 3; --L, ++lossCol) {
        const int Lm1 = L - 1;
        const int M   = B * Lm1;
        const int sJ  = S - Lm1;           // row remap jump for cur slices

        // cell 1: G = X0 @ Wih^T + b ; (h1,c1)
        gemm(stream, curA, wih_bf, nullptr, nullptr, b_lstm, 1.f, G,
             M, 2048, 512, 2048, Lm1, sJ, 0, M - 1);
        lstm_gates_kernel<<<(M * 512 + 255) / 256, 256, 0, stream>>>(G, nullptr, h1, c1, M);
        // cell 2: G = X1 @ Wih^T + h1 @ Whh^T + b ; (h2, c)
        gemm(stream, curA, wih_bf, h1, whh_bf, b_lstm, 1.f, G,
             M, 2048, 512, 2048, Lm1, sJ, 1, M - 1, 0, 0, 0);
        lstm_gates_kernel<<<(M * 512 + 255) / 256, 256, 0, stream>>>(G, c1, h2, c1, M);

        // score + argmax + gather selected
        score_fc_kernel<<<(M + 7) / 8, 256, 0, stream>>>(h2, fc_w, fc_b, sc, M);
        select_kernel<<<B, 256, 0, stream>>>(sc, h2, Lm1, selIdx, sel);

        // attention
        gemm(stream, sel, fcq_bf, nullptr, nullptr, fcq_b, 1.f, q,   B, 512, 512, 512, 0, 0, 0, B - 1);
        gemm(stream, sel, fck_bf, nullptr, nullptr, fck_b, 1.f, ksel,B, 512, 512, 512, 0, 0, 0, B - 1);
        gemm(stream, q, krel_bf, nullptr, nullptr, nullptr, invSqrtE, scores,
             B, 512, 512, 512, 0, 0, 0, B - 1);
        fixcol_kernel<<<B, 32, 0, stream>>>(q, ksel, scores, invSqrtE);
        softmax_entropy_kernel<<<B, 256, 0, stream>>>(scores, prob, pSel, lossBase, lossCol);

        // merged = prob[:, :500] @ rel_emb  (+ pSel*selected in rebuild)
        gemm(stream, prob, relT_bf, nullptr, nullptr, nullptr, 1.f, merged,
             B, 512, 512, 512, 0, 0, 0, B - 1);
        rebuild_kernel<<<B, 256, 0, stream>>>(curA, merged, sel, pSel, selIdx, curB, Lm1);

        float* t = curA; curA = curB; curB = t;
    }

    // ---- stage L == 2 : plain LSTM pair, loss col 14 = 0 ----
    {
        gemm(stream, curA, wih_bf, nullptr, nullptr, b_lstm, 1.f, G,
             B, 2048, 512, 2048, 1, 15, 0, B - 1);
        lstm_gates_kernel<<<(B * 512 + 255) / 256, 256, 0, stream>>>(G, nullptr, h1, c1, B);
        gemm(stream, curA, wih_bf, h1, whh_bf, b_lstm, 1.f, G,
             B, 2048, 512, 2048, 1, 15, 1, B - 1, 0, 0, 0);
        lstm_gates_kernel<<<(B * 512 + 255) / 256, 256, 0, stream>>>(G, c1, h2, c1, B);
        copy_h2_kernel<<<B, 256, 0, stream>>>(h2, curB);
        zero_loss_kernel<<<1, 128, 0, stream>>>(lossBase);
        float* t = curA; curA = curB; curB = t;
    }

    // ---- final attention on cur (length 1) ----
    gemm(stream, curA, fcq_bf, nullptr, nullptr, fcq_b, 1.f, q,    B, 512, 512, 512, 1, 15, 0, B - 1);
    gemm(stream, curA, fck_bf, nullptr, nullptr, fck_b, 1.f, ksel, B, 512, 512, 512, 1, 15, 0, B - 1);
    gemm(stream, q, krel_bf, nullptr, nullptr, nullptr, invSqrtE, scores,
         B, 512, 512, 512, 0, 0, 0, B - 1);
    fixcol_kernel<<<B, 32, 0, stream>>>(q, ksel, scores, invSqrtE);
    softmax_entropy_kernel<<<B, 256, 0, stream>>>(scores, nullptr, nullptr, lossBase, 15);
    copy_scores_kernel<<<(B * 501 + 255) / 256, 256, 0, stream>>>(scores, outScores);
}
